// SelfAttention_42803644072475
// MI455X (gfx1250) — matrix operand order repaired and checked by reading, not compiled, not run on hardware
//
#include <hip/hip_runtime.h>

typedef __attribute__((ext_vector_type(16))) _Float16 v16h;
typedef __attribute__((ext_vector_type(8)))  _Float16 v8h;
typedef __attribute__((ext_vector_type(8)))  float    v8f;
typedef __attribute__((ext_vector_type(4)))  float    v4f;

#define NIC 256
#define QKC 32
#define NPIX 4096

// A-fragment (16x32 f16, MxK): lane (hi = lane>>4) holds row M=lane&15,
// K = {8*hi .. 8*hi+7} then {16+8*hi .. 16+8*hi+7}  (two contiguous 16B chunks)
__device__ __forceinline__ v16h ldfragA(const _Float16* base, int hi) {
  const v8h a = *(const v8h*)(base + 8 * hi);
  const v8h b = *(const v8h*)(base + 16 + 8 * hi);
  v16h r;
#pragma unroll
  for (int i = 0; i < 8; ++i) { r[i] = a[i]; r[i + 8] = b[i]; }
  return r;
}

// B-fragment (32x16 f16, KxN): lane holds column N=lane&15,
// K = {8*hi .. 8*hi+7} then {16+8*hi .. 16+8*hi+7}  (two contiguous 16B chunks)
__device__ __forceinline__ v16h ldfragB(const _Float16* base, int hi) {
  const v8h a = *(const v8h*)(base + 8 * hi);
  const v8h b = *(const v8h*)(base + 16 + 8 * hi);
  v16h r;
#pragma unroll
  for (int i = 0; i < 8; ++i) { r[i] = a[i]; r[i + 8] = b[i]; }
  return r;
}

__global__ __launch_bounds__(256) void convert_w_kernel(
    const float* __restrict__ wq, const float* __restrict__ wk,
    const float* __restrict__ wv, _Float16* __restrict__ w16) {
  int i = blockIdx.x * 256 + threadIdx.x;
  if (i < 8192)       w16[i] = (_Float16)wq[i];
  else if (i < 16384) w16[i] = (_Float16)wk[i - 8192];
  else if (i < 81920) w16[i] = (_Float16)wv[i - 16384];
}

// Projections: Qt[b][n][32], Kt[b][n][32] (row-major per pixel),
// Vc[b][c][n] (channel-major). One WG = one batch x 64 pixel rows, 4 waves.
__global__ __launch_bounds__(128) void proj_kernel(
    const float* __restrict__ x, const _Float16* __restrict__ w16,
    const float* __restrict__ bq, const float* __restrict__ bk,
    const float* __restrict__ bv,
    _Float16* __restrict__ Qt, _Float16* __restrict__ Kt,
    _Float16* __restrict__ Vc) {
  __shared__ _Float16 xT[64 * 264];   // [n_local][c], padded stride (528B, 16B aligned)
  const int b  = blockIdx.x >> 6;
  const int n0 = (blockIdx.x & 63) * 64;
  const int tid = threadIdx.x;

  // Stage x^T tile: coalesced read along n, transposed scatter to LDS as f16
  const int tn = tid & 63;
  const int tc = tid >> 6;
  const float* xb = x + (size_t)b * NIC * NPIX;
  for (int c = tc; c < NIC; c += 2)
    xT[tn * 264 + c] = (_Float16)xb[(size_t)c * NPIX + n0 + tn];
  __syncthreads();

  const int w    = tid >> 5;
  const int lane = tid & 31;
  const int ln   = lane & 15;
  const int hi   = lane >> 4;
  const _Float16* arow = &xT[(w * 16 + ln) * 264];

  _Float16* qtb = Qt + (size_t)b * NPIX * QKC;
  _Float16* ktb = Kt + (size_t)b * NPIX * QKC;
  _Float16* vcb = Vc + (size_t)b * NIC * NPIX;

  // ---- Q and K (2 column tiles each) ----
#pragma unroll 1
  for (int qk = 0; qk < 2; ++qk) {
    const _Float16* W = w16 + qk * QKC * NIC;
    const float* bias = qk ? bk : bq;
    _Float16* dst     = qk ? ktb : qtb;
#pragma unroll 1
    for (int ot = 0; ot < 2; ++ot) {
      v8f acc = {};
#pragma unroll
      for (int k0 = 0; k0 < NIC; k0 += 32) {
        v16h A = ldfragA(arow + k0, hi);
        v16h B = ldfragB(W + (size_t)(ot * 16 + ln) * NIC + k0, hi);
        acc = __builtin_amdgcn_wmma_f32_16x16x32_f16(false, A, false, B,
                                                     (short)0, acc, false, false);
      }
      const float bb = bias[ot * 16 + ln];
#pragma unroll
      for (int g = 0; g < 8; ++g)
        dst[(size_t)(n0 + w * 16 + g + 8 * hi) * QKC + ot * 16 + ln] =
            (_Float16)(acc[g] + bb);
    }
  }

  // ---- V (16 column tiles) ----
  const _Float16* Wv = w16 + 2 * QKC * NIC;
#pragma unroll 1
  for (int ot = 0; ot < 16; ++ot) {
    v8f acc = {};
#pragma unroll
    for (int k0 = 0; k0 < NIC; k0 += 32) {
      v16h A = ldfragA(arow + k0, hi);
      v16h B = ldfragB(Wv + (size_t)(ot * 16 + ln) * NIC + k0, hi);
      acc = __builtin_amdgcn_wmma_f32_16x16x32_f16(false, A, false, B,
                                                   (short)0, acc, false, false);
    }
    const int c = ot * 16 + ln;
    const float bb = bv[c];
    v8h o;
#pragma unroll
    for (int g = 0; g < 8; ++g) o[g] = (_Float16)(acc[g] + bb);
    // D rows are consecutive pixels for a fixed channel -> 16B vector store
    *(v8h*)(vcb + (size_t)c * NPIX + n0 + w * 16 + 8 * hi) = o;
  }
}

// Flash attention + epilogue. One WG = one batch x 128 query rows, 8 waves.
// V blocks stream into LDS via double-buffered global_load_async_to_lds_b128
// (ASYNCcnt-tracked, in-order completion).
__global__ __launch_bounds__(256) void flash_kernel(
    const _Float16* __restrict__ Qt, const _Float16* __restrict__ Kt,
    const _Float16* __restrict__ Vc, const float* __restrict__ x,
    const float* __restrict__ gamma, float* __restrict__ out) {
  __shared__ _Float16 Vs2[2][NIC * 32]; // [buf][c][m_local], 2 x 16KB
  __shared__ _Float16 Ps[8 * 16 * 32];  // per-wave P tile, row-major 16x32

  const int b   = blockIdx.x >> 5;
  const int q0  = (blockIdx.x & 31) * 128;
  const int tid = threadIdx.x;
  const int w    = tid >> 5;
  const int lane = tid & 31;
  const int ln   = lane & 15;
  const int hi   = lane >> 4;

  const _Float16* qtb = Qt + (size_t)b * NPIX * QKC;
  const _Float16* ktb = Kt + (size_t)b * NPIX * QKC;
  const _Float16* vcb = Vc + (size_t)b * NIC * NPIX;

  // Issue async global->LDS copies for one 32-key V block (64B per thread,
  // thread == channel). IOFFSET applies to both global and LDS addresses.
  auto stage_async = [&](int m0, int buf) {
    unsigned long long g =
        (unsigned long long)(const void*)(vcb + (size_t)tid * NPIX + m0);
    unsigned lds = (unsigned)(unsigned long long)(const void*)&Vs2[buf][tid * 32];
    asm volatile("global_load_async_to_lds_b128 %0, %1, off"
                 :: "v"(lds), "v"(g) : "memory");
    asm volatile("global_load_async_to_lds_b128 %0, %1, off offset:16"
                 :: "v"(lds), "v"(g) : "memory");
    asm volatile("global_load_async_to_lds_b128 %0, %1, off offset:32"
                 :: "v"(lds), "v"(g) : "memory");
    asm volatile("global_load_async_to_lds_b128 %0, %1, off offset:48"
                 :: "v"(lds), "v"(g) : "memory");
  };

  const v16h Aq = ldfragA(qtb + (size_t)(q0 + w * 16 + ln) * QKC, hi);

  v8f O[16];
#pragma unroll
  for (int t = 0; t < 16; ++t) { v8f z = {}; O[t] = z; }
  float m_r[8], l_r[8];
#pragma unroll
  for (int g = 0; g < 8; ++g) { m_r[g] = -1e30f; l_r[g] = 0.0f; }

  _Float16* psw = &Ps[w * 512];

  stage_async(0, 0);   // prologue: block 0 into buffer 0

#pragma unroll 1
  for (int it = 0; it < NPIX / 32; ++it) {
    const int m0  = it * 32;
    const int buf = it & 1;

    // All waves finished reading buf^1 (iteration it-1) before we overwrite it.
    __syncthreads();
    if (it + 1 < NPIX / 32) {
      stage_async(m0 + 32, buf ^ 1);
      // 4 new copies outstanding; in-order completion => block `it` landed.
      asm volatile("s_wait_asynccnt 0x4" ::: "memory");
      // prefetch next K block while we're at it
      __builtin_prefetch(ktb + (size_t)(m0 + 32 + ln) * QKC, 0, 1);
    } else {
      asm volatile("s_wait_asynccnt 0x0" ::: "memory");
    }
    __syncthreads();
    const _Float16* vs = &Vs2[buf][0];

    // S = Q x K^T for 32 keys (two 16x16 tiles)
    v16h Bk0 = ldfragB(ktb + (size_t)(m0 + ln) * QKC, hi);
    v16h Bk1 = ldfragB(ktb + (size_t)(m0 + 16 + ln) * QKC, hi);
    v8f S0 = {}, S1 = {};
    S0 = __builtin_amdgcn_wmma_f32_16x16x32_f16(false, Aq, false, Bk0,
                                                (short)0, S0, false, false);
    S1 = __builtin_amdgcn_wmma_f32_16x16x32_f16(false, Aq, false, Bk1,
                                                (short)0, S1, false, false);

    // Online softmax: row stats live per VGPR index g within each 16-lane half
    float sc[8];
#pragma unroll
    for (int g = 0; g < 8; ++g) {
      float v = fmaxf(S0[g], S1[g]);
      v = fmaxf(v, __shfl_xor(v, 1, 32));
      v = fmaxf(v, __shfl_xor(v, 2, 32));
      v = fmaxf(v, __shfl_xor(v, 4, 32));
      v = fmaxf(v, __shfl_xor(v, 8, 32));
      float mn = fmaxf(m_r[g], v);
      sc[g] = __expf(m_r[g] - mn);
      m_r[g] = mn;
      l_r[g] *= sc[g];
    }
#pragma unroll
    for (int t = 0; t < 16; ++t)
#pragma unroll
      for (int g = 0; g < 8; ++g) O[t][g] *= sc[g];

#pragma unroll
    for (int g = 0; g < 8; ++g) {
      float p0 = __expf(S0[g] - m_r[g]);
      float p1 = __expf(S1[g] - m_r[g]);
      float s = p0 + p1;
      s += __shfl_xor(s, 1, 32);
      s += __shfl_xor(s, 2, 32);
      s += __shfl_xor(s, 4, 32);
      s += __shfl_xor(s, 8, 32);
      l_r[g] += s;
      // C/D layout -> row-major P tile in LDS (f16)
      psw[(g + 8 * hi) * 32 + ln]      = (_Float16)p0;
      psw[(g + 8 * hi) * 32 + 16 + ln] = (_Float16)p1;
    }
    asm volatile("s_wait_dscnt 0" ::: "memory");
    v16h Ap = ldfragA(&psw[ln * 32], hi);

    // O += P x V  (16 channel tiles, K = 32 local keys)
#pragma unroll
    for (int t = 0; t < 16; ++t) {
      v16h Bv = ldfragB(vs + (size_t)(t * 16 + ln) * 32, hi);
      O[t] = __builtin_amdgcn_wmma_f32_16x16x32_f16(false, Ap, false, Bv,
                                                    (short)0, O[t], false, false);
    }
  }

  // Epilogue: out = gamma[c] * (O / l) + x   (vectorized, 8 pixels per lane)
  const int nbase = q0 + w * 16 + 8 * hi;
#pragma unroll
  for (int t = 0; t < 16; ++t) {
    const int c = t * 16 + ln;
    const float g0 = gamma[c];
    const size_t base = ((size_t)b * NIC + c) * NPIX + nbase;
    v4f xa = *(const v4f*)(x + base);
    v4f xb = *(const v4f*)(x + base + 4);
    v4f oa, ob;
#pragma unroll
    for (int g = 0; g < 4; ++g) {
      oa[g] = g0 * (O[t][g]     / l_r[g])     + xa[g];
      ob[g] = g0 * (O[t][g + 4] / l_r[g + 4]) + xb[g];
    }
    *(v4f*)(out + base)     = oa;
    *(v4f*)(out + base + 4) = ob;
  }
}

extern "C" void kernel_launch(void* const* d_in, const int* in_sizes, int n_in,
                              void* d_out, int out_size, void* d_ws, size_t ws_size,
                              hipStream_t stream) {
  (void)in_sizes; (void)n_in; (void)out_size; (void)ws_size;
  const float* x     = (const float*)d_in[0];
  const float* wq    = (const float*)d_in[1];
  const float* bq    = (const float*)d_in[2];
  const float* wk    = (const float*)d_in[3];
  const float* bk    = (const float*)d_in[4];
  const float* wv    = (const float*)d_in[5];
  const float* bv    = (const float*)d_in[6];
  const float* gamma = (const float*)d_in[7];
  float* out = (float*)d_out;

  char* ws = (char*)d_ws;
  _Float16* w16 = (_Float16*)(ws);                         // 160 KB
  _Float16* Qt  = (_Float16*)(ws + 163840);                // 1 MB
  _Float16* Kt  = (_Float16*)(ws + 163840 + 1048576);      // 1 MB
  _Float16* Vc  = (_Float16*)(ws + 163840 + 2 * 1048576);  // 8 MB

  convert_w_kernel<<<320, 256, 0, stream>>>(wq, wk, wv, w16);
  proj_kernel<<<256, 128, 0, stream>>>(x, w16, bq, bk, bv, Qt, Kt, Vc);
  flash_kernel<<<128, 256, 0, stream>>>(Qt, Kt, Vc, x, gamma, out);
}
